// MVRBind_65472481460860
// MI455X (gfx1250) — compile-verified, matrix operand order checked by reading, not run on hardware
//
#include <hip/hip_runtime.h>
#include <math.h>

// ---------------------------------------------------------------------------
// Types for CDNA5 WMMA (wave32): A/B = 16 x bf16 per lane, C/D = 8 x f32.
// ---------------------------------------------------------------------------
typedef __attribute__((ext_vector_type(16))) __bf16 v16bf;
typedef __attribute__((ext_vector_type(8)))  float  v8f;

#define TPB 256
#define WPB 8   // waves per block (256 / 32)
#define CT  4   // 16-col tiles per wave -> 16x64 output per wave

static inline int cdiv(long long a, long long b) { return (int)((a + b - 1) / b); }

// ---------------------------------------------------------------------------
// Small utility kernels
// ---------------------------------------------------------------------------
__global__ void mvr_fill1d(float* __restrict__ p, float v, long long n) {
  long long i = (long long)blockIdx.x * blockDim.x + threadIdx.x;
  if (i < n) p[i] = v;
}

__global__ void mvr_fill2d(float* __restrict__ p, int rows, int cols, int ld, float v) {
  long long i = (long long)blockIdx.x * blockDim.x + threadIdx.x;
  if (i >= (long long)rows * cols) return;
  int r = (int)(i / cols), c = (int)(i - (long long)r * cols);
  p[(size_t)r * ld + c] = v;
}

__global__ void mvr_copy2d(const float* __restrict__ src, int lds,
                           float* __restrict__ dst, int ldd, int rows, int cols) {
  long long i = (long long)blockIdx.x * blockDim.x + threadIdx.x;
  if (i >= (long long)rows * cols) return;
  int r = (int)(i / cols), c = (int)(i - (long long)r * cols);
  dst[(size_t)r * ldd + c] = src[(size_t)r * lds + c];
}

__global__ void mvr_deg_add(const int* __restrict__ dst, int E, float* __restrict__ deg) {
  int e = blockIdx.x * blockDim.x + threadIdx.x;
  if (e < E) atomicAdd(&deg[dst[e]], 1.0f);
}

__global__ void mvr_rsqrt(float* __restrict__ p, int n) {
  int i = blockIdx.x * blockDim.x + threadIdx.x;
  if (i < n) p[i] = rsqrtf(fmaxf(p[i], 1.0f));
}

// ---------------------------------------------------------------------------
// WMMA GEMM:  C[M,O] = act( A[M,K] @ W[O,K]^T + bias )
// A, W are f32 in memory, converted to bf16 in registers.
// One wave computes a 16x64 output tile (4 accumulators). Per K-step all
// four B fragments are materialized in distinct registers before the four
// v_wmma_f32_16x16x32_bf16 issues, so the WMMA->VALU WAR hazard (4 co-exec
// slots, ISA 7.12.1) is paid once per iteration, not per WMMA.
// Fragment layout (CDNA5 ISA 7.12.2, 16-bit A 16x32):
//   lane 0..15  (half=0): row m = lane,    K chunks [k0..k0+7]  and [k0+16..k0+23]
//   lane 16..31 (half=1): row m = lane-16, K chunks [k0+8..k0+15] and [k0+24..k0+31]
// B (32x16, K x N) mirrors this with lane -> column n reading a row of W.
// ---------------------------------------------------------------------------
__device__ __forceinline__ v16bf mvr_cvt16(float4 a0, float4 a1, float4 a2, float4 a3) {
  v16bf f;
  f[0]  = (__bf16)a0.x; f[1]  = (__bf16)a0.y; f[2]  = (__bf16)a0.z; f[3]  = (__bf16)a0.w;
  f[4]  = (__bf16)a1.x; f[5]  = (__bf16)a1.y; f[6]  = (__bf16)a1.z; f[7]  = (__bf16)a1.w;
  f[8]  = (__bf16)a2.x; f[9]  = (__bf16)a2.y; f[10] = (__bf16)a2.z; f[11] = (__bf16)a2.w;
  f[12] = (__bf16)a3.x; f[13] = (__bf16)a3.y; f[14] = (__bf16)a3.z; f[15] = (__bf16)a3.w;
  return f;
}

__device__ __forceinline__ v16bf mvr_frag_fast(const float* __restrict__ row, int ka) {
  const float4* p = (const float4*)(row + ka);
  const float4* q = (const float4*)(row + ka + 16);
  return mvr_cvt16(p[0], p[1], q[0], q[1]);
}

// Branchless tail loader. All K used by this net are multiples of 8 and the
// two 8-float chunks are 8-aligned, so each chunk is fully in- or out-of-range:
// clamp the address (rows are >= 8 floats long), vector-load, cndmask to zero.
__device__ __forceinline__ v16bf mvr_frag_guard(const float* __restrict__ row, int ka, int K) {
  bool ok0 = (ka + 8)  <= K;
  bool ok1 = (ka + 24) <= K;
  const float4* p = (const float4*)(row + (ok0 ? ka : 0));
  const float4* q = (const float4*)(row + (ok1 ? (ka + 16) : 0));
  float4 a0 = p[0], a1 = p[1], a2 = q[0], a3 = q[1];
  a0.x = ok0 ? a0.x : 0.0f; a0.y = ok0 ? a0.y : 0.0f;
  a0.z = ok0 ? a0.z : 0.0f; a0.w = ok0 ? a0.w : 0.0f;
  a1.x = ok0 ? a1.x : 0.0f; a1.y = ok0 ? a1.y : 0.0f;
  a1.z = ok0 ? a1.z : 0.0f; a1.w = ok0 ? a1.w : 0.0f;
  a2.x = ok1 ? a2.x : 0.0f; a2.y = ok1 ? a2.y : 0.0f;
  a2.z = ok1 ? a2.z : 0.0f; a2.w = ok1 ? a2.w : 0.0f;
  a3.x = ok1 ? a3.x : 0.0f; a3.y = ok1 ? a3.y : 0.0f;
  a3.z = ok1 ? a3.z : 0.0f; a3.w = ok1 ? a3.w : 0.0f;
  return mvr_cvt16(a0, a1, a2, a3);
}

__global__ void __launch_bounds__(TPB)
mvr_gemm(const float* __restrict__ A, int lda,
         const float* __restrict__ W, int ldw,
         const float* __restrict__ bias,
         float* __restrict__ C, int ldc,
         int M, int K, int O, int act, int tn, int nTiles)
{
  int wave = threadIdx.x >> 5;
  int tile = blockIdx.x * WPB + wave;
  if (tile >= nTiles) return;          // wave-uniform: EXEC stays all-ones for WMMA
  int tm = tile / tn;
  int tc = tile - tm * tn;
  int rowBase = tm << 4, colBase = tc << 6;   // 16 rows x 64 cols per wave

  int lane = threadIdx.x & 31;
  int half = lane >> 4, ln = lane & 15;

  int ar = rowBase + ln; if (ar >= M) ar = M - 1;   // clamp: keep loads in-bounds
  const float* arow = A + (size_t)ar * lda;
  const float* brow[CT];
#pragma unroll
  for (int t = 0; t < CT; ++t) {
    int br = colBase + t * 16 + ln; if (br >= O) br = O - 1;
    brow[t] = W + (size_t)br * ldw;
  }

  v8f acc[CT];
#pragma unroll
  for (int t = 0; t < CT; ++t) acc[t] = (v8f){};

  int k0 = 0;
  for (; k0 + 32 <= K; k0 += 32) {
    int ka = k0 + half * 8;
    v16bf fa = mvr_frag_fast(arow, ka);
    v16bf fb[CT];
#pragma unroll
    for (int t = 0; t < CT; ++t) fb[t] = mvr_frag_fast(brow[t], ka);
#pragma unroll
    for (int t = 0; t < CT; ++t)
      acc[t] = __builtin_amdgcn_wmma_f32_16x16x32_bf16(false, fa, false, fb[t],
                                                       (short)0, acc[t], false, false);
  }
  if (k0 < K) {
    int ka = k0 + half * 8;
    v16bf fa = mvr_frag_guard(arow, ka, K);
    v16bf fb[CT];
#pragma unroll
    for (int t = 0; t < CT; ++t) fb[t] = mvr_frag_guard(brow[t], ka, K);
#pragma unroll
    for (int t = 0; t < CT; ++t)
      acc[t] = __builtin_amdgcn_wmma_f32_16x16x32_bf16(false, fa, false, fb[t],
                                                       (short)0, acc[t], false, false);
  }

#pragma unroll
  for (int t = 0; t < CT; ++t) {
    int col = colBase + t * 16 + ln;
    if (col < O) {
      float bv = bias ? bias[col] : 0.0f;
#pragma unroll
      for (int r = 0; r < 8; ++r) {
        int row = rowBase + r + half * 8;  // C/D layout: VGPR r, half selects M or M+8
        if (row < M) {
          float v = acc[t][r] + bv;
          if (act == 1)      v = fmaxf(v, 0.0f);
          else if (act == 2) v = 1.0f / (1.0f + __expf(-v));
          C[(size_t)row * ldc + col] = v;
        }
      }
    }
  }
}

// ---------------------------------------------------------------------------
// GCN scatter:  out[dst] += xw[src] * dis[src]*dis[dst]   (per edge, 136 feats)
// ---------------------------------------------------------------------------
__global__ void mvr_gcn_scatter(const float* __restrict__ xw,
                                const int* __restrict__ src, const int* __restrict__ dst,
                                const float* __restrict__ dis,
                                float* __restrict__ out, int ldo, int E, int F)
{
  int chunks = F >> 2;
  long long idx = (long long)blockIdx.x * blockDim.x + threadIdx.x;
  if (idx >= (long long)E * chunks) return;
  int e = (int)(idx / chunks);
  int c = (int)(idx - (long long)e * chunks) << 2;
  int s = src[e], d = dst[e];
  float nrm = dis[s] * dis[d];
  float4 v = *(const float4*)(xw + (size_t)s * F + c);
  float* o = out + (size_t)d * ldo + c;
  atomicAdd(o + 0, v.x * nrm);
  atomicAdd(o + 1, v.y * nrm);
  atomicAdd(o + 2, v.z * nrm);
  atomicAdd(o + 3, v.w * nrm);
}

// out = relu(out + xw * dis^2 + b)   -- self-loop + bias + activation
__global__ void mvr_gcn_finalize(const float* __restrict__ xw, const float* __restrict__ dis,
                                 const float* __restrict__ b,
                                 float* __restrict__ out, int ldo, int n, int F)
{
  long long idx = (long long)blockIdx.x * blockDim.x + threadIdx.x;
  if (idx >= (long long)n * F) return;
  int i = (int)(idx / F), f = (int)(idx - (long long)i * F);
  float d = dis[i];
  float v = out[(size_t)i * ldo + f] + xw[(size_t)i * F + f] * d * d + b[f];
  out[(size_t)i * ldo + f] = fmaxf(v, 0.0f);
}

// ---------------------------------------------------------------------------
// LayerNorm of (a + b) over C features, wave-per-row (wave32 shuffle reduce)
// ---------------------------------------------------------------------------
__global__ void __launch_bounds__(TPB)
mvr_ln(const float* __restrict__ a, const float* __restrict__ b,
       const float* __restrict__ g, const float* __restrict__ beta,
       float* __restrict__ out, int rows, int C, int ld)
{
  int row = blockIdx.x * WPB + (threadIdx.x >> 5);
  if (row >= rows) return;
  int lane = threadIdx.x & 31;
  const float* pa = a + (size_t)row * ld;
  const float* pb = b + (size_t)row * ld;

  float s = 0.0f;
  for (int c = lane; c < C; c += 32) s += pa[c] + pb[c];
  for (int o = 16; o; o >>= 1) s += __shfl_xor(s, o, 32);
  float mean = s / (float)C;

  float vs = 0.0f;
  for (int c = lane; c < C; c += 32) { float t = pa[c] + pb[c] - mean; vs += t * t; }
  for (int o = 16; o; o >>= 1) vs += __shfl_xor(vs, o, 32);
  float rstd = rsqrtf(vs / (float)C + 1e-5f);

  float* po = out + (size_t)row * ld;
  for (int c = lane; c < C; c += 32) {
    float t = pa[c] + pb[c];
    po[c] = (t - mean) * rstd * g[c] + beta[c];
  }
}

// ---------------------------------------------------------------------------
// 4-token attention head: per node, softmax(QK^T/sqrt(128)) @ V -> [512]
// ---------------------------------------------------------------------------
__global__ void __launch_bounds__(TPB)
mvr_attn4(const float* __restrict__ Q, const float* __restrict__ Km,
          const float* __restrict__ V, float* __restrict__ out, int n)
{
  int node = blockIdx.x * WPB + (threadIdx.x >> 5);
  if (node >= n) return;
  int lane = threadIdx.x & 31;
  const float* q = Q  + (size_t)node * 512;
  const float* k = Km + (size_t)node * 512;
  const float* v = V  + (size_t)node * 512;
  float sc[4][4];
#pragma unroll
  for (int qt = 0; qt < 4; ++qt)
#pragma unroll
    for (int kt = 0; kt < 4; ++kt) {
      float p = 0.0f;
      for (int d = lane; d < 128; d += 32) p += q[qt * 128 + d] * k[kt * 128 + d];
      for (int o = 16; o; o >>= 1) p += __shfl_xor(p, o, 32);
      sc[qt][kt] = p * 0.08838834764831845f;  // 1/sqrt(128)
    }
  float* po = out + (size_t)node * 512;
#pragma unroll
  for (int qt = 0; qt < 4; ++qt) {
    float m = fmaxf(fmaxf(sc[qt][0], sc[qt][1]), fmaxf(sc[qt][2], sc[qt][3]));
    float e0 = __expf(sc[qt][0] - m), e1 = __expf(sc[qt][1] - m);
    float e2 = __expf(sc[qt][2] - m), e3 = __expf(sc[qt][3] - m);
    float inv = 1.0f / (e0 + e1 + e2 + e3);
    float a0 = e0 * inv, a1 = e1 * inv, a2 = e2 * inv, a3 = e3 * inv;
    for (int d = lane; d < 128; d += 32)
      po[qt * 128 + d] = a0 * v[d] + a1 * v[128 + d] + a2 * v[256 + d] + a3 * v[384 + d];
  }
}

// ---------------------------------------------------------------------------
// Host-side orchestration helpers
// ---------------------------------------------------------------------------
static void gemm(hipStream_t s, const float* A, int lda, const float* W, int ldw,
                 const float* bias, float* C, int ldc, int M, int K, int O, int act)
{
  int tm = (M + 15) >> 4, tn = (O + 63) >> 6;   // 16x64 per wave
  int tiles = tm * tn;
  mvr_gemm<<<cdiv(tiles, WPB), TPB, 0, s>>>(A, lda, W, ldw, bias, C, ldc,
                                            M, K, O, act, tn, tiles);
}

static void graph_norm(hipStream_t s, const int* dst, int E, float* dis, int N)
{
  mvr_fill1d<<<cdiv(N, TPB), TPB, 0, s>>>(dis, 1.0f, N);  // self-loop degree
  mvr_deg_add<<<cdiv(E, TPB), TPB, 0, s>>>(dst, E, dis);
  mvr_rsqrt<<<cdiv(N, TPB), TPB, 0, s>>>(dis, N);
}

static void gcn(hipStream_t s, const float* xin, int lda, int K,
                const float* W, const float* b,
                const int* src, const int* dst, const float* dis, int E,
                float* outBase, int ldo, float* xw, int N)
{
  gemm(s, xin, lda, W, K, nullptr, xw, 136, N, K, 136, 0);
  mvr_fill2d<<<cdiv((long long)N * 136, TPB), TPB, 0, s>>>(outBase, N, 136, ldo, 0.0f);
  mvr_gcn_scatter<<<cdiv((long long)E * 34, TPB), TPB, 0, s>>>(xw, src, dst, dis,
                                                               outBase, ldo, E, 136);
  mvr_gcn_finalize<<<cdiv((long long)N * 136, TPB), TPB, 0, s>>>(xw, dis, b,
                                                                 outBase, ldo, N, 136);
}

static void tlayer(hipStream_t s, const float* in, float* out,
                   const float* const* P, int base,
                   float* ta, float* tb, float* tff, int N)
{
  const float* Wv = P[base + 0] + 2 * 408 * 408;  // V rows of in_proj weight
  const float* bv = P[base + 1] + 2 * 408;
  const float* Wo = P[base + 2]; const float* bo  = P[base + 3];
  const float* W1 = P[base + 4]; const float* b1  = P[base + 5];
  const float* W2 = P[base + 6]; const float* b2  = P[base + 7];
  const float* g1 = P[base + 8]; const float* be1 = P[base + 9];
  const float* g2 = P[base + 10]; const float* be2 = P[base + 11];
  gemm(s, in, 408, Wv, 408, bv, ta, 408, N, 408, 408, 0);     // V projection
  gemm(s, ta, 408, Wo, 408, bo, tb, 408, N, 408, 408, 0);     // out_proj
  mvr_ln<<<cdiv(N, WPB), TPB, 0, s>>>(in, tb, g1, be1, ta, N, 408, 408);
  gemm(s, ta, 408, W1, 408, b1, tff, 256, N, 408, 256, 1);    // FFN up + relu
  gemm(s, tff, 256, W2, 256, b2, tb, 408, N, 256, 408, 0);    // FFN down
  mvr_ln<<<cdiv(N, WPB), TPB, 0, s>>>(ta, tb, g2, be2, out, N, 408, 408);
}

extern "C" void kernel_launch(void* const* d_in, const int* in_sizes, int n_in,
                              void* d_out, int out_size, void* d_ws, size_t ws_size,
                              hipStream_t stream)
{
  const float* x  = (const float*)d_in[0];
  const int*   eM = (const int*)d_in[1];
  const int*   eS = (const int*)d_in[2];
  const int*   eF = (const int*)d_in[3];
  const int N  = in_sizes[0] / 136;
  const int EM = in_sizes[1] / 2;
  const int ES = in_sizes[2] / 2;
  const int EF = in_sizes[3] / 2;

  // Params in setup_inputs() insertion order (offset from d_in[4]):
  const float* P[60];
  for (int i = 0; i < 60; ++i) P[i] = (const float*)d_in[4 + i];
  // 0:conv1_W 1:b 2:conv2_W 3:b 4:conv3_W 5:b
  // 6..11: conv{1,2,3}_se_{W,b}   12..17: conv{1,2,3}_f_{W,b}
  // 18..29: t0 (in_w,in_b,out_W,out_b,l1_W,l1_b,l2_W,l2_b,ln1_g,ln1_b,ln2_g,ln2_b)
  // 30..41: t1   42..47: linc{1,2,3}_{W,b}   48..53: q/k/v {W,b}
  // 54..59: lin4_{W,b} lin1_{W,b} lin3_{W,b}

  // --- workspace bump allocator ---
  char* ws = (char*)d_ws;
  size_t off = 0;
  auto alloc = [&](size_t nflt) -> float* {
    float* p = (float*)(ws + off);
    off = (off + nflt * sizeof(float) + 255) & ~(size_t)255;
    return p;
  };
  float* disM = alloc(N);
  float* disS = alloc(N);
  float* disF = alloc(N);
  float* xw   = alloc((size_t)N * 136);
  float* cat1 = alloc((size_t)N * 408);
  float* cat2 = alloc((size_t)N * 408);
  float* cat3 = alloc((size_t)N * 408);
  float* xb1  = alloc((size_t)N * 408);
  float* xb2  = alloc((size_t)N * 408);
  float* xb3  = alloc((size_t)N * 408);
  float* ta   = alloc((size_t)N * 408);
  float* tb   = alloc((size_t)N * 408);
  float* tff  = alloc((size_t)N * 256);
  float* xs   = alloc((size_t)N * 544);   // [N,4,136] token stack
  float* Qb   = alloc((size_t)N * 512);
  float* Kb   = alloc((size_t)N * 512);
  float* Vb   = alloc((size_t)N * 512);
  float* x5   = alloc((size_t)N * 512);
  float* h1   = alloc((size_t)N * 256);
  float* h2   = alloc((size_t)N * 128);

  // --- GCN normalization terms for the three graphs ---
  graph_norm(stream, eM + EM, EM, disM, N);
  graph_norm(stream, eS + ES, ES, disS, N);
  graph_norm(stream, eF + EF, EF, disF, N);

  // --- structural (sed) branch -> column slice [136:272) of cat buffers ---
  gcn(stream, x,          136, 136, P[6],  P[7],  eS, eS + ES, disS, ES, cat1 + 136, 408, xw, N);
  gcn(stream, cat1 + 136, 408, 136, P[8],  P[9],  eS, eS + ES, disS, ES, cat2 + 136, 408, xw, N);
  gcn(stream, cat2 + 136, 408, 136, P[10], P[11], eS, eS + ES, disS, ES, cat3 + 136, 408, xw, N);

  // --- window branch -> column slice [272:408) ---
  gcn(stream, x,          136, 136, P[12], P[13], eF, eF + EF, disF, EF, cat1 + 272, 408, xw, N);
  gcn(stream, cat1 + 272, 408, 136, P[14], P[15], eF, eF + EF, disF, EF, cat2 + 272, 408, xw, N);
  gcn(stream, cat2 + 272, 408, 136, P[16], P[17], eF, eF + EF, disF, EF, cat3 + 272, 408, xw, N);

  // --- main branch interleaved with the shared transformer ---
  gcn(stream, x,   136, 136, P[0], P[1], eM, eM + EM, disM, EM, cat1, 408, xw, N);
  tlayer(stream, cat1, xb1, P, 18, ta, tb, tff, N);
  tlayer(stream, xb1,  xb1, P, 30, ta, tb, tff, N);

  gcn(stream, xb1, 408, 408, P[2], P[3], eM, eM + EM, disM, EM, cat2, 408, xw, N);
  tlayer(stream, cat2, xb2, P, 18, ta, tb, tff, N);
  tlayer(stream, xb2,  xb2, P, 30, ta, tb, tff, N);

  gcn(stream, xb2, 408, 408, P[4], P[5], eM, eM + EM, disM, EM, cat3, 408, xw, N);
  tlayer(stream, cat3, xb3, P, 18, ta, tb, tff, N);
  tlayer(stream, xb3,  xb3, P, 30, ta, tb, tff, N);

  // --- token stack xs = [x, relu(linc1(x1)), relu(linc2(x2)), relu(linc3(x3))] ---
  mvr_copy2d<<<cdiv((long long)N * 136, TPB), TPB, 0, stream>>>(x, 136, xs, 544, N, 136);
  gemm(stream, xb1, 408, P[42], 408, P[43], xs + 136, 544, N, 408, 136, 1);
  gemm(stream, xb2, 408, P[44], 408, P[45], xs + 272, 544, N, 408, 136, 1);
  gemm(stream, xb3, 408, P[46], 408, P[47], xs + 408, 544, N, 408, 136, 1);

  // --- 4-token attention: xs is [4N,136] row-major ---
  gemm(stream, xs, 136, P[48], 136, P[49], Qb, 128, 4 * N, 136, 128, 0);
  gemm(stream, xs, 136, P[50], 136, P[51], Kb, 128, 4 * N, 136, 128, 0);
  gemm(stream, xs, 136, P[52], 136, P[53], Vb, 128, 4 * N, 136, 128, 0);
  mvr_attn4<<<cdiv(N, WPB), TPB, 0, stream>>>(Qb, Kb, Vb, x5, N);

  // --- MLP head ---
  gemm(stream, x5, 512, P[54], 512, P[55], h1, 256, N, 512, 256, 1);
  gemm(stream, h1, 256, P[56], 256, P[57], h2, 128, N, 256, 128, 1);
  gemm(stream, h2, 128, P[58], 128, P[59], (float*)d_out, 1, N, 128, 1, 2);
}
